// CliqueGNN_62466004353626
// MI455X (gfx1250) — matrix operand order repaired and checked by reading, not compiled, not run on hardware
//
#include <hip/hip_runtime.h>
#include <hip/hip_bf16.h>

// ---------------------------------------------------------------------------
// CliqueGNN forward for MI455X (gfx1250, wave32, WMMA).
// Big E x 128 edge tensors live in fp16; edge GEMMs use v_wmma_f32_16x16x32_f16
// with weights transposed to [N][K] fp16 and staged in LDS.
// ---------------------------------------------------------------------------

typedef _Float16 v16h __attribute__((ext_vector_type(16)));
typedef float    v8f  __attribute__((ext_vector_type(8)));

union AFrag {
    v16h  h;
    uint4 q[2];   // two 16-byte chunks = 16 halves
};

#define EPSBN 1e-5f

// ---------------------------------------------------------------------------
// Small utility kernels
// ---------------------------------------------------------------------------
__global__ __launch_bounds__(256) void k_f32_to_f16(const float* __restrict__ a,
                                                    _Float16* __restrict__ b, int n) {
    int i = blockIdx.x * 256 + threadIdx.x;
    if (i < n) b[i] = (_Float16)a[i];
}

// w is [K,128] row-major fp32 -> out is [128][K] fp16 (N-major, K contiguous)
__global__ __launch_bounds__(256) void k_transpose_w(const float* __restrict__ w,
                                                     _Float16* __restrict__ o, int K) {
    int i = blockIdx.x * 256 + threadIdx.x;
    if (i < K * 128) {
        int k = i >> 7, n = i & 127;
        o[n * K + k] = (_Float16)w[i];
    }
}

__global__ __launch_bounds__(256) void k_deg(const int* __restrict__ dst, float* __restrict__ deg, int E) {
    int i = blockIdx.x * 256 + threadIdx.x;
    if (i < E) atomicAdd(&deg[dst[i]], 1.0f);
}

__global__ __launch_bounds__(256) void k_dis(const float* __restrict__ deg, float* __restrict__ dis, int V) {
    int i = blockIdx.x * 256 + threadIdx.x;
    if (i < V) dis[i] = rsqrtf(deg[i] + 1.0f);
}

// ef = edge_attr(E,3) @ edge_W(3,128) + edge_b   -> fp16 [E,128]
__global__ __launch_bounds__(256) void k_edge_embed(const float* __restrict__ attr,
                                                    const float* __restrict__ W,
                                                    const float* __restrict__ b,
                                                    _Float16* __restrict__ ef, int E) {
    int i = blockIdx.x * 256 + threadIdx.x;        // E*32 threads, 4 channels each
    if (i >= E * 32) return;
    int e  = i >> 5;
    int c0 = (i & 31) * 4;
    float a0 = attr[e * 3 + 0], a1 = attr[e * 3 + 1], a2 = attr[e * 3 + 2];
#pragma unroll
    for (int j = 0; j < 4; j++) {
        int c = c0 + j;
        float v = a0 * W[c] + a1 * W[128 + c] + a2 * W[256 + c] + b[c];
        ef[(size_t)e * 128 + c] = (_Float16)v;
    }
}

// ---------------------------------------------------------------------------
// GCN path (small: 512 nodes) — fp32 scalar
// ---------------------------------------------------------------------------
__global__ __launch_bounds__(256) void k_gcn_h(const float* __restrict__ x,
                                               const float* __restrict__ W,
                                               const float* __restrict__ b,
                                               const float* __restrict__ dis,
                                               float* __restrict__ h,
                                               float* __restrict__ agg, int V) {
    int i = blockIdx.x * 256 + threadIdx.x;        // V*128
    if (i >= V * 128) return;
    int v = i >> 7, c = i & 127;
    float s = 0.f;
    for (int k = 0; k < 128; k++) s += x[(size_t)v * 128 + k] * W[(size_t)k * 128 + c];
    h[i] = s;
    float dv = dis[v];
    agg[i] = dv * dv * s + b[c];                   // self-loop + bias init
}

__global__ __launch_bounds__(256) void k_gcn_scatter(const int* __restrict__ src,
                                                     const int* __restrict__ dst,
                                                     const float* __restrict__ dis,
                                                     const float* __restrict__ h,
                                                     float* __restrict__ agg, int E) {
    int i = blockIdx.x * 256 + threadIdx.x;        // E*32, 4 channels each
    if (i >= E * 32) return;
    int e = i >> 5, c0 = (i & 31) * 4;
    int s = src[e], d = dst[e];
    float nm = dis[s] * dis[d];
#pragma unroll
    for (int j = 0; j < 4; j++)
        atomicAdd(&agg[(size_t)d * 128 + c0 + j], nm * h[(size_t)s * 128 + c0 + j]);
}

// BN (batch stats over V rows) + ReLU, one block per channel
__global__ __launch_bounds__(256) void k_bn_relu_node(const float* __restrict__ agg,
                                                      const float* __restrict__ g,
                                                      const float* __restrict__ beta,
                                                      float* __restrict__ xout, int V) {
    int c = blockIdx.x;                            // 0..127
    __shared__ float sS[256], sQ[256];
    float s = 0.f, q = 0.f;
    for (int v = threadIdx.x; v < V; v += 256) {
        float t = agg[(size_t)v * 128 + c];
        s += t; q += t * t;
    }
    sS[threadIdx.x] = s; sQ[threadIdx.x] = q;
    __syncthreads();
    for (int o = 128; o > 0; o >>= 1) {
        if ((int)threadIdx.x < o) { sS[threadIdx.x] += sS[threadIdx.x + o]; sQ[threadIdx.x] += sQ[threadIdx.x + o]; }
        __syncthreads();
    }
    float m = sS[0] / V;
    float var = sQ[0] / V - m * m;
    float inv = rsqrtf(var + EPSBN);
    float gc = g[c], bc = beta[c];
    for (int v = threadIdx.x; v < V; v += 256) {
        float t = gc * (agg[(size_t)v * 128 + c] - m) * inv + bc;
        xout[(size_t)v * 128 + c] = t > 0.f ? t : 0.f;
    }
}

// ---------------------------------------------------------------------------
// WMMA edge GEMM: out(E,128) fp16 = A(E,KDIM) @ Wt + bias
//   MODE 0: A = [x16[src] | x16[dst]]   (KDIM=256, gather)
//   MODE 1: A = ef                      (KDIM=128)
//   MODE 2: A = [nf | ee]               (KDIM=256)
// Wt is [128][KDIM] fp16 (N-major); staged into LDS; each wave owns 16 edges.
// ---------------------------------------------------------------------------
template <int KDIM, int MODE>
__global__ __launch_bounds__(256) void k_edge_wmma(const _Float16* __restrict__ a0,
                                                   const _Float16* __restrict__ a1,
                                                   const int* __restrict__ src,
                                                   const int* __restrict__ dst,
                                                   const _Float16* __restrict__ wT,
                                                   const float* __restrict__ bias,
                                                   _Float16* __restrict__ out, int E) {
    __shared__ __align__(16) _Float16 sW[128 * KDIM];   // <= 64KB

    // cooperative stage of transposed weights into LDS (b128 copies)
    {
        const int cnt = (128 * KDIM) / 8;               // uint4 chunks
        const uint4* gsrc = (const uint4*)wT;
        uint4* ldst = (uint4*)sW;
        for (int i = threadIdx.x; i < cnt; i += 256) ldst[i] = gsrc[i];
    }
    __syncthreads();

    const int lane = threadIdx.x & 31;
    const int wave = threadIdx.x >> 5;
    const int m    = lane & 15;
    const int khi  = lane >> 4;                          // 0: lanes 0-15, 1: lanes 16-31
    const int nTiles = E >> 4;

    for (int tile = blockIdx.x * 8 + wave; tile < nTiles; tile += gridDim.x * 8) {
        const int e = tile * 16 + m;
        const _Float16 *ps, *pd;
        if (MODE == 0) {
            ps = a0 + (size_t)src[e] * 128;
            pd = a0 + (size_t)dst[e] * 128;
        } else if (MODE == 1) {
            ps = a0 + (size_t)e * 128; pd = ps;
        } else {
            ps = a0 + (size_t)e * 128;
            pd = a1 + (size_t)e * 128;
        }

        // Preload all A fragments for this 16-edge strip (reused over 8 col-tiles)
        AFrag a[KDIM / 32];
#pragma unroll
        for (int ks = 0; ks < KDIM / 32; ks++) {
            int k0 = ks * 32 + 8 * khi;                  // 8 contiguous halves
            int k1 = k0 + 16;                            // 8 contiguous halves
            const _Float16* p0 = (k0 < 128) ? (ps + k0) : (pd + (k0 - 128));
            const _Float16* p1 = (k1 < 128) ? (ps + k1) : (pd + (k1 - 128));
            a[ks].q[0] = *(const uint4*)p0;
            a[ks].q[1] = *(const uint4*)p1;
        }

        const int ncol = lane & 15;
#pragma unroll
        for (int ct = 0; ct < 8; ct++) {
            v8f acc = {0.f, 0.f, 0.f, 0.f, 0.f, 0.f, 0.f, 0.f};
            const int n = ct * 16 + ncol;
            const _Float16* wrow = &sW[(size_t)n * KDIM + 8 * khi];
#pragma unroll
            for (int ks = 0; ks < KDIM / 32; ks++) {
                AFrag b;
                const _Float16* bp = wrow + ks * 32;     // 8 halves at k0, 8 halves at k0+16
                b.q[0] = *(const uint4*)bp;
                b.q[1] = *(const uint4*)(bp + 16);
                acc = __builtin_amdgcn_wmma_f32_16x16x32_f16(
                    false, a[ks].h, false, b.h, (short)0, acc, false, false);
            }
            const float bv = bias[n];
#pragma unroll
            for (int r = 0; r < 8; r++) {
                int row = r + 8 * khi;                   // C layout: VGPR r -> M = r + 8*hi
                float v = acc[r] + bv;
                out[(size_t)(tile * 16 + row) * 128 + n] = (_Float16)v;
            }
        }
    }
}

// ---------------------------------------------------------------------------
// Column stats over E rows (sum / sumsq per channel) — fp16 input
// ---------------------------------------------------------------------------
__global__ __launch_bounds__(256) void k_col_stats(const _Float16* __restrict__ d, int rows,
                                                   float* __restrict__ sum, float* __restrict__ sq) {
    __shared__ float sS[128], sQ[128];
    if (threadIdx.x < 128) { sS[threadIdx.x] = 0.f; sQ[threadIdx.x] = 0.f; }
    __syncthreads();
    const int c = threadIdx.x & 127;
    const int rstep = (gridDim.x * 256) >> 7;
    float s = 0.f, q = 0.f;
    for (int r = (blockIdx.x * 256 + threadIdx.x) >> 7; r < rows; r += rstep) {
        float t = (float)d[(size_t)r * 128 + c];
        s += t; q += t * t;
    }
    atomicAdd(&sS[c], s);
    atomicAdd(&sQ[c], q);
    __syncthreads();
    if (threadIdx.x < 128) {
        atomicAdd(&sum[threadIdx.x], sS[threadIdx.x]);
        atomicAdd(&sq[threadIdx.x],  sQ[threadIdx.x]);
    }
}

__global__ __launch_bounds__(256) void k_bn_relu_edge(const _Float16* __restrict__ comb,
                                                      const float* __restrict__ sum,
                                                      const float* __restrict__ sq,
                                                      const float* __restrict__ g,
                                                      const float* __restrict__ beta,
                                                      _Float16* __restrict__ ef, int rows) {
    size_t n = (size_t)rows * 128;
    for (size_t i = (size_t)blockIdx.x * 256 + threadIdx.x; i < n; i += (size_t)gridDim.x * 256) {
        int c = (int)(i & 127);
        float m = sum[c] / rows;
        float var = sq[c] / rows - m * m;
        float inv = rsqrtf(var + EPSBN);
        float t = g[c] * ((float)comb[i] - m) * inv + beta[c];
        ef[i] = (_Float16)(t > 0.f ? t : 0.f);
    }
}

// ---------------------------------------------------------------------------
// Policy head
// ---------------------------------------------------------------------------
__global__ __launch_bounds__(256) void k_policy_scores(const _Float16* __restrict__ ef,
                                                       const float* __restrict__ pW,
                                                       const float* __restrict__ pb,
                                                       const int* __restrict__ src,
                                                       const int* __restrict__ dst,
                                                       float* __restrict__ scores, int E, int V) {
    int e = blockIdx.x * 256 + threadIdx.x;
    if (e >= E) return;
    float s = 0.f;
    const _Float16* row = ef + (size_t)e * 128;
    for (int c = 0; c < 128; c++) s += (float)row[c] * pW[c];
    s += pb[0];
    int a = src[e], b = dst[e];
    if (a < b) {
        int tri = a * (2 * V - a - 1) / 2 + (b - a - 1);
        scores[tri] = s;
    }
}

__global__ __launch_bounds__(256) void k_block_max(const float* __restrict__ x, int n, float* __restrict__ part) {
    __shared__ float s[256];
    float m = -3.4e38f;
    for (int i = blockIdx.x * 256 + threadIdx.x; i < n; i += gridDim.x * 256) m = fmaxf(m, x[i]);
    s[threadIdx.x] = m; __syncthreads();
    for (int o = 128; o > 0; o >>= 1) {
        if ((int)threadIdx.x < o) s[threadIdx.x] = fmaxf(s[threadIdx.x], s[threadIdx.x + o]);
        __syncthreads();
    }
    if (!threadIdx.x) part[blockIdx.x] = s[0];
}

__global__ __launch_bounds__(128) void k_final_max(const float* __restrict__ part, int nb, float* __restrict__ g) {
    __shared__ float s[128];
    s[threadIdx.x] = ((int)threadIdx.x < nb) ? part[threadIdx.x] : -3.4e38f;
    __syncthreads();
    for (int o = 64; o > 0; o >>= 1) {
        if ((int)threadIdx.x < o) s[threadIdx.x] = fmaxf(s[threadIdx.x], s[threadIdx.x + o]);
        __syncthreads();
    }
    if (!threadIdx.x) g[0] = s[0];
}

__global__ __launch_bounds__(256) void k_block_sumexp(const float* __restrict__ x, int n,
                                                      const float* __restrict__ gmax, float* __restrict__ part) {
    __shared__ float s[256];
    float gm = gmax[0];
    float a = 0.f;
    for (int i = blockIdx.x * 256 + threadIdx.x; i < n; i += gridDim.x * 256) a += expf(x[i] - gm);
    s[threadIdx.x] = a; __syncthreads();
    for (int o = 128; o > 0; o >>= 1) {
        if ((int)threadIdx.x < o) s[threadIdx.x] += s[threadIdx.x + o];
        __syncthreads();
    }
    if (!threadIdx.x) part[blockIdx.x] = s[0];
}

__global__ __launch_bounds__(128) void k_final_sum(const float* __restrict__ part, int nb, float* __restrict__ g) {
    __shared__ float s[128];
    s[threadIdx.x] = ((int)threadIdx.x < nb) ? part[threadIdx.x] : 0.f;
    __syncthreads();
    for (int o = 64; o > 0; o >>= 1) {
        if ((int)threadIdx.x < o) s[threadIdx.x] += s[threadIdx.x + o];
        __syncthreads();
    }
    if (!threadIdx.x) g[0] = s[0];
}

__global__ __launch_bounds__(256) void k_softmax_out(const float* __restrict__ x, int n,
                                                     const float* __restrict__ gmax,
                                                     const float* __restrict__ gsum,
                                                     float* __restrict__ out) {
    int i = blockIdx.x * 256 + threadIdx.x;
    if (i < n) out[i] = expf(x[i] - gmax[0]) / gsum[0];
}

// ---------------------------------------------------------------------------
// Value head: mean-pool nodes -> relu(linear 128->64) -> tanh(linear 64->1)
// ---------------------------------------------------------------------------
__global__ __launch_bounds__(128) void k_value(const float* __restrict__ x,
                                               const float* __restrict__ W1, const float* __restrict__ b1,
                                               const float* __restrict__ W2, const float* __restrict__ b2,
                                               float* __restrict__ out, int V) {
    __shared__ float pooled[128];
    __shared__ float hid[64];
    int c = threadIdx.x;
    float s = 0.f;
    for (int v = 0; v < V; v++) s += x[(size_t)v * 128 + c];
    pooled[c] = s / V;
    __syncthreads();
    if (c < 64) {
        float h = 0.f;
        for (int k = 0; k < 128; k++) h += pooled[k] * W1[k * 64 + c];
        h += b1[c];
        hid[c] = h > 0.f ? h : 0.f;
    }
    __syncthreads();
    if (c == 0) {
        float v = 0.f;
        for (int k = 0; k < 64; k++) v += hid[k] * W2[k];
        out[0] = tanhf(v + b2[0]);
    }
}

// ---------------------------------------------------------------------------
// Host launcher
// ---------------------------------------------------------------------------
extern "C" void kernel_launch(void* const* d_in, const int* in_sizes, int n_in,
                              void* d_out, int out_size, void* d_ws, size_t ws_size,
                              hipStream_t stream) {
    const int E = in_sizes[0] / 2;
    const int V = in_sizes[2] / 128;
    const int E_UND = out_size - 1;
    const int L = 2;

    const int*   src       = (const int*)d_in[0];
    const int*   dst       = src + E;
    const float* edge_attr = (const float*)d_in[1];
    const float* node_emb  = (const float*)d_in[2];
    const float* edge_W    = (const float*)d_in[3];
    const float* edge_b    = (const float*)d_in[4];
    const float* gcn_W[2]   = {(const float*)d_in[5],  (const float*)d_in[6]};
    const float* gcn_b[2]   = {(const float*)d_in[7],  (const float*)d_in[8]};
    const float* gcn_g[2]   = {(const float*)d_in[9],  (const float*)d_in[10]};
    const float* gcn_bt[2]  = {(const float*)d_in[11], (const float*)d_in[12]};
    const float* eb_eW[2]   = {(const float*)d_in[13], (const float*)d_in[14]};
    const float* eb_eb[2]   = {(const float*)d_in[15], (const float*)d_in[16]};
    const float* eb_nW[2]   = {(const float*)d_in[17], (const float*)d_in[18]};
    const float* eb_nb[2]   = {(const float*)d_in[19], (const float*)d_in[20]};
    const float* eb_cW[2]   = {(const float*)d_in[21], (const float*)d_in[22]};
    const float* eb_cb[2]   = {(const float*)d_in[23], (const float*)d_in[24]};
    const float* eb_g[2]    = {(const float*)d_in[25], (const float*)d_in[26]};
    const float* eb_bt[2]   = {(const float*)d_in[27], (const float*)d_in[28]};
    const float* pol_W  = (const float*)d_in[29];
    const float* pol_b  = (const float*)d_in[30];
    const float* val_W1 = (const float*)d_in[31];
    const float* val_b1 = (const float*)d_in[32];
    const float* val_W2 = (const float*)d_in[33];
    const float* val_b2 = (const float*)d_in[34];

    // -------- workspace carve ----------
    char* wsp = (char*)d_ws;
    size_t off = 0;
    auto carve = [&](size_t bytes) -> void* {
        void* p = wsp + off;
        off += (bytes + 255) & ~(size_t)255;
        return p;
    };
    _Float16* ef   = (_Float16*)carve((size_t)E * 128 * 2);
    _Float16* nf   = (_Float16*)carve((size_t)E * 128 * 2);
    _Float16* ee   = (_Float16*)carve((size_t)E * 128 * 2);
    _Float16* comb = (_Float16*)carve((size_t)E * 128 * 2);
    float* xA   = (float*)carve((size_t)V * 128 * 4);
    float* xB   = (float*)carve((size_t)V * 128 * 4);
    _Float16* x16 = (_Float16*)carve((size_t)V * 128 * 2);
    float* hbuf = (float*)carve((size_t)V * 128 * 4);
    float* agg  = (float*)carve((size_t)V * 128 * 4);
    float* deg  = (float*)carve((size_t)V * 4);
    float* dis  = (float*)carve((size_t)V * 4);
    _Float16* nWT[2] = {(_Float16*)carve(256 * 128 * 2), (_Float16*)carve(256 * 128 * 2)};
    _Float16* eWT[2] = {(_Float16*)carve(128 * 128 * 2), (_Float16*)carve(128 * 128 * 2)};
    _Float16* cWT[2] = {(_Float16*)carve(256 * 128 * 2), (_Float16*)carve(256 * 128 * 2)};
    float* bnSum = (float*)carve(128 * 4);
    float* bnSq  = (float*)carve(128 * 4);
    float* scores = (float*)carve((size_t)E_UND * 4);
    float* part   = (float*)carve(256 * 4);
    float* gmax   = (float*)carve(4);
    float* gsum   = (float*)carve(4);
    (void)ws_size; (void)n_in;

    float* pol_out = (float*)d_out;           // [E_UND] policy, then [1] value
    float* val_out = pol_out + E_UND;

    // -------- weight prep (fp16 transposed) ----------
    for (int l = 0; l < L; l++) {
        k_transpose_w<<<dim3((256 * 128 + 255) / 256), dim3(256), 0, stream>>>(eb_nW[l], nWT[l], 256);
        k_transpose_w<<<dim3((128 * 128 + 255) / 256), dim3(256), 0, stream>>>(eb_eW[l], eWT[l], 128);
        k_transpose_w<<<dim3((256 * 128 + 255) / 256), dim3(256), 0, stream>>>(eb_cW[l], cWT[l], 256);
    }

    // -------- degrees / dis ----------
    hipMemsetAsync(deg, 0, (size_t)V * 4, stream);
    k_deg<<<dim3((E + 255) / 256), dim3(256), 0, stream>>>(dst, deg, E);
    k_dis<<<dim3((V + 255) / 256), dim3(256), 0, stream>>>(deg, dis, V);

    // -------- x init, ef init ----------
    hipMemcpyAsync(xA, node_emb, (size_t)V * 128 * 4, hipMemcpyDeviceToDevice, stream);
    k_edge_embed<<<dim3((E * 32 + 255) / 256), dim3(256), 0, stream>>>(edge_attr, edge_W, edge_b, ef, E);

    float* x_cur = xA;
    float* x_nxt = xB;

    for (int l = 0; l < L; l++) {
        // old x -> fp16 for EdgeBlock gathers
        k_f32_to_f16<<<dim3((V * 128 + 255) / 256), dim3(256), 0, stream>>>(x_cur, x16, V * 128);

        // ---- GNNBlock (fp32, small) ----
        k_gcn_h<<<dim3((V * 128 + 255) / 256), dim3(256), 0, stream>>>(x_cur, gcn_W[l], gcn_b[l], dis, hbuf, agg, V);
        k_gcn_scatter<<<dim3((E * 32 + 255) / 256), dim3(256), 0, stream>>>(src, dst, dis, hbuf, agg, E);
        k_bn_relu_node<<<dim3(128), dim3(256), 0, stream>>>(agg, gcn_g[l], gcn_bt[l], x_nxt, V);

        // ---- EdgeBlock (WMMA, fp16 storage, fp32 accum) ----
        k_edge_wmma<256, 0><<<dim3(256), dim3(256), 0, stream>>>(x16, (const _Float16*)nullptr, src, dst,
                                                                 nWT[l], eb_nb[l], nf, E);
        k_edge_wmma<128, 1><<<dim3(256), dim3(256), 0, stream>>>(ef, (const _Float16*)nullptr, src, dst,
                                                                 eWT[l], eb_eb[l], ee, E);
        k_edge_wmma<256, 2><<<dim3(256), dim3(256), 0, stream>>>(nf, ee, src, dst,
                                                                 cWT[l], eb_cb[l], comb, E);

        // BN over E rows + ReLU -> new ef
        hipMemsetAsync(bnSum, 0, 128 * 4, stream);
        hipMemsetAsync(bnSq, 0, 128 * 4, stream);
        k_col_stats<<<dim3(512), dim3(256), 0, stream>>>(comb, E, bnSum, bnSq);
        k_bn_relu_edge<<<dim3(2048), dim3(256), 0, stream>>>(comb, bnSum, bnSq, eb_g[l], eb_bt[l], ef, E);

        // swap x buffers
        float* t = x_cur; x_cur = x_nxt; x_nxt = t;
    }

    // -------- policy head ----------
    k_policy_scores<<<dim3((E + 255) / 256), dim3(256), 0, stream>>>(ef, pol_W, pol_b, src, dst, scores, E, V);
    k_block_max<<<dim3(128), dim3(256), 0, stream>>>(scores, E_UND, part);
    k_final_max<<<dim3(1), dim3(128), 0, stream>>>(part, 128, gmax);
    k_block_sumexp<<<dim3(128), dim3(256), 0, stream>>>(scores, E_UND, gmax, part);
    k_final_sum<<<dim3(1), dim3(128), 0, stream>>>(part, 128, gsum);
    k_softmax_out<<<dim3((E_UND + 255) / 256), dim3(256), 0, stream>>>(scores, E_UND, gmax, gsum, pol_out);

    // -------- value head ----------
    k_value<<<dim3(1), dim3(128), 0, stream>>>(x_cur, val_W1, val_b1, val_W2, val_b2, val_out, V);
}
